// VectorQuantizer_31172872634801
// MI455X (gfx1250) — matrix operand order repaired and checked by reading, not compiled, not run on hardware
//
#include <hip/hip_runtime.h>

// ---------------------------------------------------------------------------
// VQ-VAE vector quantizer for MI455X (gfx1250, wave32, WMMA).
//   z:        (32768 x 256) f32 rows (reshape of (32,256,32,32))
//   codebook: (8192  x 256) f32
// Outputs (concatenated f32): z_q_st (8388608), loss (1), indices-as-f32 (32768)
//
// d = ||e||^2 - 2 z.e (||z||^2 is per-row constant -> irrelevant for argmin).
// GEMM uses bf16x3 hi/lo split on v_wmma_f32_16x16x32_bf16 (~f32 accuracy at
// bf16 matrix rates). Codebook is pre-converted ONCE to bf16 hi/lo images in
// workspace; the hot kernel streams those through a double-buffered LDS
// pipeline (regs prefetch of chunk c+1 overlaps the 48 WMMAs of chunk c).
// ---------------------------------------------------------------------------

#define D_DIM   256
#define K_CODES 8192
#define N_ROWS  32768
#define ZELEMS  (N_ROWS * D_DIM)   /* 8388608 */
#define LOSS_OFF ZELEMS
#define IDX_OFF  (ZELEMS + 1)

#define CHUNK_CODES 32             /* codes per LDS stage            */
#define NSTEPS      (K_CODES / CHUNK_CODES)
#define LDS_PITCH   264            /* 256 + 8 pad -> conflict-free   */
#define HALF_U16    (CHUNK_CODES * LDS_PITCH)        /* 8448 ushorts */
#define BUF_U16     (2 * HALF_U16)                   /* hi + lo      */
#define SMEM_U16    (2 * BUF_U16)                    /* двух buffers */
#define SMEM_BYTES  (SMEM_U16 * 2 + 2 * CHUNK_CODES * 4)  /* +eNorm  */

typedef __bf16          v16bf  __attribute__((ext_vector_type(16)));
typedef unsigned short  v16u16 __attribute__((ext_vector_type(16)));
typedef unsigned short  v8u16  __attribute__((ext_vector_type(8)));
typedef float           v8f    __attribute__((ext_vector_type(8)));

static __device__ __forceinline__ unsigned short f2bfu(float x) {
  unsigned u = __builtin_bit_cast(unsigned, x);
  u += 0x7fffu + ((u >> 16) & 1u);          // round-to-nearest-even
  return (unsigned short)(u >> 16);
}
static __device__ __forceinline__ float bfu2f(unsigned short h) {
  unsigned u = ((unsigned)h) << 16;
  return __builtin_bit_cast(float, u);
}

// ---------------------------------------------------------------- kernel 1 --
// One block per code: eNorm[k] = sum e^2 (exact f32), and bf16 hi/lo images.
__global__ __launch_bounds__(256) void vq_prep_kernel(
    const float* __restrict__ codebook, float* __restrict__ eNorm,
    unsigned short* __restrict__ cbHi, unsigned short* __restrict__ cbLo) {
  const int code = blockIdx.x;
  const int t = threadIdx.x;
  const size_t o = (size_t)code * D_DIM + t;
  float x = codebook[o];
  unsigned short hb = f2bfu(x);
  cbHi[o] = hb;
  cbLo[o] = f2bfu(x - bfu2f(hb));
  float v = x * x;
#pragma unroll
  for (int m = 16; m >= 1; m >>= 1) v += __shfl_xor(v, m, 32);
  __shared__ float s[8];
  if ((t & 31) == 0) s[t >> 5] = v;
  __syncthreads();
  if (t == 0) {
    float tot = 0.f;
#pragma unroll
    for (int i = 0; i < 8; ++i) tot += s[i];
    eNorm[code] = tot;
  }
}

// ---------------------------------------------------------------- kernel 2 --
// Fused GEMM + argmin. 8 waves/block, 16 rows/wave (128 rows/block).
__global__ __launch_bounds__(256) void vq_argmin_kernel(
    const float* __restrict__ z, const unsigned short* __restrict__ cbHi,
    const unsigned short* __restrict__ cbLo, const float* __restrict__ eNorm,
    int* __restrict__ idxOut, float* __restrict__ idxOutF) {
  extern __shared__ unsigned short smem[];          // double-buffered stage
  float* enlds = (float*)(smem + SMEM_U16);

  const int tid   = threadIdx.x;
  const int lane  = tid & 31;
  const int wave  = tid >> 5;
  const int nloc  = lane & 15;          // N column within 16x16 tile
  const int khalf = lane >> 4;          // which K half this lane holds

  const int rowBase = blockIdx.x * 128 + wave * 16;
  const float* zrow = z + (size_t)(rowBase + nloc) * D_DIM;
  const int koff = khalf * 16;          // A: lanes16-31 hold K {16..31}

  // ---- A fragments (hi/lo bf16), register resident for the whole scan ----
  v16u16 ahi[8], alo[8];
#pragma unroll
  for (int ch = 0; ch < 8; ++ch) {
    v16u16 h, l;
    const int base = ch * 32 + koff;
#pragma unroll
    for (int e = 0; e < 8; ++e) {
      float x = zrow[base + e];               // K = koff+e
      unsigned short hb = f2bfu(x);
      h[e] = hb;  l[e] = f2bfu(x - bfu2f(hb));
      float y = zrow[base + 8 + e];           // K = koff+8+e
      unsigned short hb2 = f2bfu(y);
      h[8 + e] = hb2;  l[8 + e] = f2bfu(y - bfu2f(hb2));
    }
    ahi[ch] = h;  alo[ch] = l;
  }

  float minV[8];  int minI[8];
#pragma unroll
  for (int r = 0; r < 8; ++r) { minV[r] = 3.4e38f; minI[r] = 0; }

  // ---- staging helpers (pure b128 traffic, no arithmetic) ----
  const int sCode = (tid * 8) >> 8;           // 8 ushorts per thread per it
  const int sK    = (tid * 8) & 255;
  v8u16 nh[4], nl[4];
  float enr = 0.f;

  // prologue: chunk 0 -> regs -> LDS buffer 0
#pragma unroll
  for (int it = 0; it < 4; ++it) {
    const size_t g = (size_t)(it * 8 + sCode) * D_DIM + sK;   // it*8 codes apart
    nh[it] = *(const v8u16*)&cbHi[g];
    nl[it] = *(const v8u16*)&cbLo[g];
  }
  if (tid < CHUNK_CODES) enr = eNorm[tid];
#pragma unroll
  for (int it = 0; it < 4; ++it) {
    const int o = (it * 8 + sCode) * LDS_PITCH + sK;
    *(v8u16*)&smem[o]            = nh[it];
    *(v8u16*)&smem[HALF_U16 + o] = nl[it];
  }
  if (tid < CHUNK_CODES) enlds[tid] = enr;
  __syncthreads();

  for (int step = 0; step < NSTEPS; ++step) {
    const int cur = step & 1;
    unsigned short* bufHi = smem + cur * BUF_U16;
    unsigned short* bufLo = bufHi + HALF_U16;
    const float*    bufEn = enlds + cur * CHUNK_CODES;

    // issue next chunk's global loads (latency hidden under the WMMAs)
    if (step + 1 < NSTEPS) {
      const size_t cb0 = (size_t)(step + 1) * CHUNK_CODES;
#pragma unroll
      for (int it = 0; it < 4; ++it) {
        const size_t g = (cb0 + it * 8 + sCode) * D_DIM + sK;
        nh[it] = *(const v8u16*)&cbHi[g];
        nl[it] = *(const v8u16*)&cbLo[g];
      }
      if (tid < CHUNK_CODES) enr = eNorm[cb0 + tid];
    }

    // ---- 16x16 code tiles: 8 K-chunks x 3 bf16 WMMAs (hi/lo split) ----
#pragma unroll
    for (int sub = 0; sub < CHUNK_CODES / 16; ++sub) {
      v8f acc = {};
#pragma unroll
      for (int ch = 0; ch < 8; ++ch) {
        const int off = (sub * 16 + nloc) * LDS_PITCH + ch * 32 + khalf * 16;
        v8u16 h0 = *(const v8u16*)&bufHi[off];
        v8u16 h1 = *(const v8u16*)&bufHi[off + 8];
        v8u16 l0 = *(const v8u16*)&bufLo[off];
        v8u16 l1 = *(const v8u16*)&bufLo[off + 8];
        v16u16 bhu = __builtin_shufflevector(h0, h1, 0,1,2,3,4,5,6,7,8,9,10,11,12,13,14,15);
        v16u16 blu = __builtin_shufflevector(l0, l1, 0,1,2,3,4,5,6,7,8,9,10,11,12,13,14,15);
        v16bf Ah = __builtin_bit_cast(v16bf, ahi[ch]);
        v16bf Al = __builtin_bit_cast(v16bf, alo[ch]);
        v16bf Bh = __builtin_bit_cast(v16bf, bhu);
        v16bf Bl = __builtin_bit_cast(v16bf, blu);
        acc = __builtin_amdgcn_wmma_f32_16x16x32_bf16(false, Ah, false, Bh, (short)0, acc, false, false);
        acc = __builtin_amdgcn_wmma_f32_16x16x32_bf16(false, Al, false, Bh, (short)0, acc, false, false);
        acc = __builtin_amdgcn_wmma_f32_16x16x32_bf16(false, Ah, false, Bl, (short)0, acc, false, false);
      }
      const float en  = bufEn[sub * 16 + nloc];
      const int  cidx = step * CHUNK_CODES + sub * 16 + nloc;
#pragma unroll
      for (int r = 0; r < 8; ++r) {
        float d = en - 2.0f * acc[r];
        if (d < minV[r]) { minV[r] = d; minI[r] = cidx; }   // first-min tie rule
      }
    }

    // store next chunk into the other buffer (nobody reads it this step)
    if (step + 1 < NSTEPS) {
      unsigned short* nHi = smem + (1 - cur) * BUF_U16;
#pragma unroll
      for (int it = 0; it < 4; ++it) {
        const int o = (it * 8 + sCode) * LDS_PITCH + sK;
        *(v8u16*)&nHi[o]            = nh[it];
        *(v8u16*)&nHi[HALF_U16 + o] = nl[it];
      }
      if (tid < CHUNK_CODES) enlds[(1 - cur) * CHUNK_CODES + tid] = enr;
    }
    __syncthreads();
  }

  // ---- argmin across the 16 lanes of each N-group (C layout:
  //      lanes 0-15 -> rows r, lanes 16-31 -> rows r+8) ----
#pragma unroll
  for (int r = 0; r < 8; ++r) {
    float v = minV[r];  int i = minI[r];
#pragma unroll
    for (int m = 1; m <= 8; m <<= 1) {
      float ov = __shfl_xor(v, m, 32);
      int   oi = __shfl_xor(i, m, 32);
      if (ov < v || (ov == v && oi < i)) { v = ov; i = oi; }
    }
    if (nloc == 0) {
      int orow = rowBase + khalf * 8 + r;
      idxOut[orow]  = i;
      idxOutF[orow] = (float)i;
    }
  }
}

// ---------------------------------------------------------------- kernel 3 --
// Gather z_q = codebook[idx], emit per-block partial sum of (z_q - z)^2.
__global__ __launch_bounds__(256) void vq_gather_kernel(
    const float* __restrict__ z, const float* __restrict__ codebook,
    const int* __restrict__ idx, float* __restrict__ zq,
    float* __restrict__ partials) {
  const int wave = threadIdx.x >> 5;
  const int lane = threadIdx.x & 31;
  const int row = blockIdx.x * 8 + wave;
  const int id = idx[row];
  const float* e  = codebook + (size_t)id * D_DIM;
  const float* zr = z  + (size_t)row * D_DIM;
  float*       o  = zq + (size_t)row * D_DIM;
  float acc = 0.f;
#pragma unroll
  for (int j = 0; j < 8; ++j) {
    int k = lane + j * 32;
    float q = e[k];
    float d = q - zr[k];
    o[k] = q;
    acc += d * d;
  }
#pragma unroll
  for (int m = 16; m >= 1; m >>= 1) acc += __shfl_xor(acc, m, 32);
  __shared__ float s[8];
  if (lane == 0) s[wave] = acc;
  __syncthreads();
  if (threadIdx.x == 0) {
    float t = 0.f;
#pragma unroll
    for (int i = 0; i < 8; ++i) t += s[i];   // fixed order -> deterministic
    partials[blockIdx.x] = t;
  }
}

// ---------------------------------------------------------------- kernel 4 --
// loss = 1.25 * sum(partials) / ZELEMS   (deterministic single-block reduce)
__global__ __launch_bounds__(256) void vq_loss_kernel(
    const float* __restrict__ partials, float* __restrict__ lossOut) {
  float acc = 0.f;
  for (int i = threadIdx.x; i < N_ROWS / 8; i += 256) acc += partials[i];
#pragma unroll
  for (int m = 16; m >= 1; m >>= 1) acc += __shfl_xor(acc, m, 32);
  __shared__ float s[8];
  if ((threadIdx.x & 31) == 0) s[threadIdx.x >> 5] = acc;
  __syncthreads();
  if (threadIdx.x == 0) {
    float tot = 0.f;
#pragma unroll
    for (int i = 0; i < 8; ++i) tot += s[i];
    lossOut[0] = tot * (1.25f / (float)ZELEMS);
  }
}

// ------------------------------------------------------------------ launch --
extern "C" void kernel_launch(void* const* d_in, const int* in_sizes, int n_in,
                              void* d_out, int out_size, void* d_ws, size_t ws_size,
                              hipStream_t stream) {
  const float* z        = (const float*)d_in[0];   // 8388608 f32
  const float* codebook = (const float*)d_in[1];   // 2097152 f32
  float* out = (float*)d_out;

  // workspace layout (4-byte units for the f32/int region):
  float* eNorm    = (float*)d_ws;                       // [0, 8192)
  int*   idxWs    = (int*)d_ws + K_CODES;               // [8192, 40960)
  float* partials = (float*)d_ws + K_CODES + N_ROWS;    // [40960, 45056)
  unsigned short* cbHi = (unsigned short*)((float*)d_ws + 45056);  // 4 MB
  unsigned short* cbLo = cbHi + (size_t)K_CODES * D_DIM;           // 4 MB

  vq_prep_kernel  <<<K_CODES,      256, 0,          stream>>>(codebook, eNorm,
                                                              cbHi, cbLo);
  vq_argmin_kernel<<<N_ROWS / 128, 256, SMEM_BYTES, stream>>>(z, cbHi, cbLo,
                                                              eNorm, idxWs,
                                                              out + IDX_OFF);
  vq_gather_kernel<<<N_ROWS / 8,   256, 0,          stream>>>(z, codebook,
                                                              idxWs, out,
                                                              partials);
  vq_loss_kernel  <<<1,            256, 0,          stream>>>(partials,
                                                              out + LOSS_OFF);
}
